// MoELayer_81630148428171
// MI455X (gfx1250) — matrix operand order repaired and checked by reading, not compiled, not run on hardware
//
#include <hip/hip_runtime.h>
#include <cstdint>

#define HDIM 1024
#define NEXP 8
#define FF   4096
#define NTOK 8192                    // B*S = 4*2048
#define RMAX (2*NTOK + 16*NEXP)      // 16512 rows (top-2 slots, per-expert pad to 16)
#define BK   32
#define BN   512                     // block N tile: 8 waves x 4 wmma tiles

typedef __attribute__((ext_vector_type(16))) __bf16 v16bf;
typedef __attribute__((ext_vector_type(8)))  float  v8f;

__device__ __forceinline__ uint16_t bfbits(float f) {
  union { float f; uint32_t u; } v; v.f = f;
  uint32_t r = v.u + 0x7FFFu + ((v.u >> 16) & 1u);   // round-to-nearest-even
  return (uint16_t)(r >> 16);
}
__device__ __forceinline__ __bf16 f2bf(float f) {
  union { uint16_t s; __bf16 b; } o; o.s = bfbits(f);
  return o.b;
}
__device__ __forceinline__ float gelu_exact(float v) {
  return 0.5f * v * (1.0f + erff(v * 0.70710678118654752f));
}

// ---------------- one-time fp32 -> bf16 conversion (8 elts/thread) ----------------

__global__ __launch_bounds__(256) void moe_cvt_bf16(const float* __restrict__ in,
                                                    __bf16* __restrict__ out) {
  size_t base = ((size_t)blockIdx.x * 256 + threadIdx.x) * 8;
  const float4* p = (const float4*)(in + base);
  float4 a = p[0], b = p[1];
  union { uint16_t u[8]; uint4 q; } o;
  o.u[0] = bfbits(a.x); o.u[1] = bfbits(a.y); o.u[2] = bfbits(a.z); o.u[3] = bfbits(a.w);
  o.u[4] = bfbits(b.x); o.u[5] = bfbits(b.y); o.u[6] = bfbits(b.z); o.u[7] = bfbits(b.w);
  *(uint4*)(out + base) = o.q;
}

// ---------------- routing ----------------

__global__ __launch_bounds__(256) void moe_init(int* counts, int* list_token,
                                                int* list_dest) {
  int i = blockIdx.x * 256 + threadIdx.x;
  if (i < NEXP) counts[i] = 0;
  if (i < RMAX) { list_token[i] = 0; list_dest[i] = -1; }
}

__global__ __launch_bounds__(256) void moe_router(
    const float* __restrict__ x, const float* __restrict__ rw,
    const float* __restrict__ rb, float* __restrict__ top_w,
    int* __restrict__ top_idx, int* __restrict__ counts) {
  __shared__ float sW[HDIM * NEXP];
  for (int i = threadIdx.x; i < HDIM * NEXP; i += 256) sW[i] = rw[i];
  __syncthreads();
  int t = blockIdx.x * 256 + threadIdx.x;
  const float* xt = x + (size_t)t * HDIM;
  float acc[NEXP];
#pragma unroll
  for (int e = 0; e < NEXP; ++e) acc[e] = rb[e];
  for (int h = 0; h < HDIM; ++h) {
    float xv = xt[h];
#pragma unroll
    for (int e = 0; e < NEXP; ++e) acc[e] += xv * sW[h * NEXP + e];
  }
  float m = acc[0];
#pragma unroll
  for (int e = 1; e < NEXP; ++e) m = fmaxf(m, acc[e]);
  float p[NEXP];
#pragma unroll
  for (int e = 0; e < NEXP; ++e) p[e] = __expf(acc[e] - m);
  int i0 = 0; float v0 = p[0];
#pragma unroll
  for (int e = 1; e < NEXP; ++e) if (p[e] > v0) { v0 = p[e]; i0 = e; }
  int i1 = (i0 == 0) ? 1 : 0; float v1 = p[i1];
#pragma unroll
  for (int e = 0; e < NEXP; ++e) if (e != i0 && p[e] > v1) { v1 = p[e]; i1 = e; }
  float d = v0 + v1;                 // softmax denominator cancels in the ratio
  top_w[2 * t] = v0 / d; top_w[2 * t + 1] = v1 / d;
  top_idx[2 * t] = i0;   top_idx[2 * t + 1] = i1;
  atomicAdd(&counts[i0], 1); atomicAdd(&counts[i1], 1);
}

__global__ void moe_prefix(const int* counts, int* row_base, int* fill) {
  if (threadIdx.x == 0 && blockIdx.x == 0) {
    int pref = 0;
    for (int e = 0; e < NEXP; ++e) {
      row_base[e] = pref; fill[e] = 0;
      pref += ((counts[e] + 15) >> 4) << 4;
    }
    row_base[NEXP] = pref;
  }
}

__global__ __launch_bounds__(256) void moe_scatter(
    const int* __restrict__ top_idx, const int* __restrict__ row_base,
    int* __restrict__ fill, int* __restrict__ list_token,
    int* __restrict__ list_dest) {
  int t = blockIdx.x * 256 + threadIdx.x;
  if (t >= NTOK) return;
#pragma unroll
  for (int s = 0; s < 2; ++s) {
    int e = top_idx[2 * t + s];
    int pos = atomicAdd(&fill[e], 1);
    int row = row_base[e] + pos;
    list_token[row] = t;
    list_dest[row] = 2 * t + s;
  }
}

// ---------------- grouped GEMM 1: hmid = gelu(xbf @ w1bf[e] + b1[e]) ----------------

__global__ __launch_bounds__(256) void moe_gemm1(
    const __bf16* __restrict__ xbf, const __bf16* __restrict__ w1bf,
    const float* __restrict__ bias1, const int* __restrict__ row_base,
    const int* __restrict__ list_token, __bf16* __restrict__ hmid) {
  int row0 = blockIdx.x * 16;
  if (row0 >= row_base[NEXP]) return;
  int e = 0;
  while (row0 >= row_base[e + 1]) ++e;
  int colBase = blockIdx.y * BN;

  __shared__ __bf16 sA[16][BK];         // 1 KB
  __shared__ __bf16 sB[BN][BK + 8];     // [n][k] transposed, 40 KB
  __shared__ int sTok[16];
  if (threadIdx.x < 16) sTok[threadIdx.x] = list_token[row0 + threadIdx.x];
  __syncthreads();

  const int lane = threadIdx.x & 31;
  const int wave = threadIdx.x >> 5;
  const int n0 = wave * 64;             // 4 wmma n-tiles per wave
  const __bf16* wptr = w1bf + (size_t)e * HDIM * FF + colBase;

  const int kg = (threadIdx.x & 7) * 4;   // this thread's 4 k-rows
  const int ng = (threadIdx.x >> 3) * 16; // this thread's 16 n-cols

  v8f acc[4] = {{}, {}, {}, {}};

  for (int kk = 0; kk < HDIM; kk += BK) {
    { // stage A: 16x32 bf16, one b32 per thread
      int r = threadIdx.x >> 4;
      int c = (threadIdx.x & 15) * 2;
      *(uint32_t*)&sA[r][c] = *(const uint32_t*)(xbf + (size_t)sTok[r] * HDIM + kk + c);
    }
    { // stage B: 4k x 16n subtile per thread, repack to [n][k], b64 stores
      const __bf16* wrow = wptr + (size_t)(kk + kg) * FF + ng;
      union { uint4 q; uint16_t u[8]; } row[4][2];
#pragma unroll
      for (int i = 0; i < 4; ++i) {
        const uint4* p = (const uint4*)(wrow + (size_t)i * FF);
        row[i][0].q = p[0]; row[i][1].q = p[1];
      }
      if (kk + BK < HDIM) __builtin_prefetch(wrow + (size_t)BK * FF, 0, 1);
#pragma unroll
      for (int j = 0; j < 16; ++j) {
        union { uint16_t u[4]; uint2 d; } pk;
        int h = j >> 3, jj = j & 7;
        pk.u[0] = row[0][h].u[jj]; pk.u[1] = row[1][h].u[jj];
        pk.u[2] = row[2][h].u[jj]; pk.u[3] = row[3][h].u[jj];
        *(uint2*)&sB[ng + j][kg] = pk.d;
      }
    }
    __syncthreads();
    union { v16bf v; uint4 q[2]; } a;
    {
      int r = lane & 15, ko = (lane >> 4) * 8;   // ISA 16-bit A layout
      a.q[0] = *(const uint4*)&sA[r][ko];
      a.q[1] = *(const uint4*)&sA[r][16 + ko];
    }
    int n = lane & 15, ko2 = (lane >> 4) * 8;    // ISA 16-bit B layout
#pragma unroll
    for (int tix = 0; tix < 4; ++tix) {
      union { v16bf v; uint4 q[2]; } b;
      const __bf16* bp = &sB[n0 + tix * 16 + n][ko2];
      b.q[0] = *(const uint4*)bp;
      b.q[1] = *(const uint4*)(bp + 16);
      acc[tix] = __builtin_amdgcn_wmma_f32_16x16x32_bf16(
          false, a.v, false, b.v, (short)0, acc[tix], false, false);
    }
    __syncthreads();
  }

  const float* be = bias1 + (size_t)e * FF;
  int n = lane & 15;
  int rlo = (lane >> 4) * 8;                     // C/D: vgpr r -> M = r + 8*(lane>=16)
#pragma unroll
  for (int tix = 0; tix < 4; ++tix) {
#pragma unroll
    for (int r = 0; r < 8; ++r) {
      int c = colBase + n0 + tix * 16 + n;
      size_t grow = (size_t)(row0 + rlo + r);
      hmid[grow * FF + c] = f2bf(gelu_exact(acc[tix][r] + be[c]));
    }
  }
}

// ---------------- grouped GEMM 2: y = hmid @ w2bf[e] + b2[e] ----------------

__global__ __launch_bounds__(256) void moe_gemm2(
    const __bf16* __restrict__ hmid, const __bf16* __restrict__ w2bf,
    const float* __restrict__ bias2, const int* __restrict__ row_base,
    const int* __restrict__ list_dest, float* __restrict__ ybuf) {
  int row0 = blockIdx.x * 16;
  if (row0 >= row_base[NEXP]) return;
  int e = 0;
  while (row0 >= row_base[e + 1]) ++e;
  int colBase = blockIdx.y * BN;

  __shared__ __bf16 sA[16][BK];
  __shared__ __bf16 sB[BN][BK + 8];
  __shared__ int sDst[16];
  if (threadIdx.x < 16) sDst[threadIdx.x] = list_dest[row0 + threadIdx.x];
  __syncthreads();

  const int lane = threadIdx.x & 31;
  const int wave = threadIdx.x >> 5;
  const int n0 = wave * 64;
  const __bf16* wptr = w2bf + (size_t)e * FF * HDIM + colBase;

  const int kg = (threadIdx.x & 7) * 4;
  const int ng = (threadIdx.x >> 3) * 16;

  v8f acc[4] = {{}, {}, {}, {}};

  for (int kk = 0; kk < FF; kk += BK) {
    {
      int r = threadIdx.x >> 4;
      int c = (threadIdx.x & 15) * 2;
      *(uint32_t*)&sA[r][c] = *(const uint32_t*)(hmid + (size_t)(row0 + r) * FF + kk + c);
    }
    {
      const __bf16* wrow = wptr + (size_t)(kk + kg) * HDIM + ng;
      union { uint4 q; uint16_t u[8]; } row[4][2];
#pragma unroll
      for (int i = 0; i < 4; ++i) {
        const uint4* p = (const uint4*)(wrow + (size_t)i * HDIM);
        row[i][0].q = p[0]; row[i][1].q = p[1];
      }
      if (kk + BK < FF) __builtin_prefetch(wrow + (size_t)BK * HDIM, 0, 1);
#pragma unroll
      for (int j = 0; j < 16; ++j) {
        union { uint16_t u[4]; uint2 d; } pk;
        int h = j >> 3, jj = j & 7;
        pk.u[0] = row[0][h].u[jj]; pk.u[1] = row[1][h].u[jj];
        pk.u[2] = row[2][h].u[jj]; pk.u[3] = row[3][h].u[jj];
        *(uint2*)&sB[ng + j][kg] = pk.d;
      }
    }
    __syncthreads();
    union { v16bf v; uint4 q[2]; } a;
    {
      int r = lane & 15, ko = (lane >> 4) * 8;
      a.q[0] = *(const uint4*)&sA[r][ko];
      a.q[1] = *(const uint4*)&sA[r][16 + ko];
    }
    int n = lane & 15, ko2 = (lane >> 4) * 8;
#pragma unroll
    for (int tix = 0; tix < 4; ++tix) {
      union { v16bf v; uint4 q[2]; } b;
      const __bf16* bp = &sB[n0 + tix * 16 + n][ko2];
      b.q[0] = *(const uint4*)bp;
      b.q[1] = *(const uint4*)(bp + 16);
      acc[tix] = __builtin_amdgcn_wmma_f32_16x16x32_bf16(
          false, a.v, false, b.v, (short)0, acc[tix], false, false);
    }
    __syncthreads();
  }

  const float* be = bias2 + (size_t)e * HDIM;
  int n = lane & 15;
  int rlo = (lane >> 4) * 8;
#pragma unroll
  for (int tix = 0; tix < 4; ++tix) {
#pragma unroll
    for (int r = 0; r < 8; ++r) {
      int dst = sDst[rlo + r];
      if (dst < 0) continue;                   // padding row
      int c = colBase + n0 + tix * 16 + n;
      ybuf[(size_t)dst * HDIM + c] = acc[tix][r] + be[c];
    }
  }
}

// ---------------- combine ----------------

__global__ __launch_bounds__(256) void moe_combine(
    const float* __restrict__ ybuf, const float* __restrict__ top_w,
    float* __restrict__ out) {
  size_t i = (size_t)blockIdx.x * 256 + threadIdx.x;
  int t = (int)(i >> 10);
  int h = (int)(i & (HDIM - 1));
  out[i] = top_w[2 * t]     * ybuf[(size_t)(2 * t)     * HDIM + h]
         + top_w[2 * t + 1] * ybuf[(size_t)(2 * t + 1) * HDIM + h];
}

// ---------------- host ----------------

extern "C" void kernel_launch(void* const* d_in, const int* in_sizes, int n_in,
                              void* d_out, int out_size, void* d_ws, size_t ws_size,
                              hipStream_t stream) {
  const float* x  = (const float*)d_in[0];
  const float* rw = (const float*)d_in[1];
  const float* rb = (const float*)d_in[2];
  const float* w1 = (const float*)d_in[3];
  const float* b1 = (const float*)d_in[4];
  const float* w2 = (const float*)d_in[5];
  const float* b2 = (const float*)d_in[6];
  float* out = (float*)d_out;

  char* ws = (char*)d_ws;
  const size_t OFF_TOPW = 0;
  const size_t OFF_TOPI = OFF_TOPW + (size_t)NTOK * 2 * 4;
  const size_t OFF_CNT  = OFF_TOPI + (size_t)NTOK * 2 * 4;
  const size_t OFF_FILL = OFF_CNT + 256;
  const size_t OFF_RB   = OFF_FILL + 256;
  const size_t OFF_LTOK = OFF_RB + 256;
  const size_t OFF_LDST = OFF_LTOK + (size_t)RMAX * 4;
  size_t off = OFF_LDST + (size_t)RMAX * 4;
  off = (off + 255) & ~(size_t)255;
  const size_t OFF_XBF  = off;                                     // NTOK*H bf16
  const size_t OFF_W1BF = OFF_XBF  + (size_t)NTOK * HDIM * 2;      // E*H*FF bf16
  const size_t OFF_W2BF = OFF_W1BF + (size_t)NEXP * HDIM * FF * 2; // E*FF*H bf16
  const size_t OFF_HMID = OFF_W2BF + (size_t)NEXP * FF * HDIM * 2; // RMAX*FF bf16
  const size_t OFF_YBUF = OFF_HMID + (size_t)RMAX * FF * 2;        // 2*NTOK*H f32

  float* top_w      = (float*)(ws + OFF_TOPW);
  int*   top_idx    = (int*)  (ws + OFF_TOPI);
  int*   counts     = (int*)  (ws + OFF_CNT);
  int*   fill       = (int*)  (ws + OFF_FILL);
  int*   row_base   = (int*)  (ws + OFF_RB);
  int*   list_token = (int*)  (ws + OFF_LTOK);
  int*   list_dest  = (int*)  (ws + OFF_LDST);
  __bf16* xbf       = (__bf16*)(ws + OFF_XBF);
  __bf16* w1bf      = (__bf16*)(ws + OFF_W1BF);
  __bf16* w2bf      = (__bf16*)(ws + OFF_W2BF);
  __bf16* hmid      = (__bf16*)(ws + OFF_HMID);
  float* ybuf       = (float*)(ws + OFF_YBUF);

  // one-time bf16 conversion of activations + weights
  moe_cvt_bf16<<<((size_t)NTOK * HDIM) / 2048, 256, 0, stream>>>(x, xbf);
  moe_cvt_bf16<<<((size_t)NEXP * HDIM * FF) / 2048, 256, 0, stream>>>(w1, w1bf);
  moe_cvt_bf16<<<((size_t)NEXP * FF * HDIM) / 2048, 256, 0, stream>>>(w2, w2bf);

  moe_init   <<<(RMAX + 255) / 256, 256, 0, stream>>>(counts, list_token, list_dest);
  moe_router <<<NTOK / 256, 256, 0, stream>>>(x, rw, rb, top_w, top_idx, counts);
  moe_prefix <<<1, 1, 0, stream>>>(counts, row_base, fill);
  moe_scatter<<<NTOK / 256, 256, 0, stream>>>(top_idx, row_base, fill, list_token, list_dest);

  dim3 g1(RMAX / 16, FF / BN);      // 1032 x 8
  moe_gemm1<<<g1, 256, 0, stream>>>(xbf, w1bf, b1, row_base, list_token, hmid);

  dim3 g2(RMAX / 16, HDIM / BN);    // 1032 x 2
  moe_gemm2<<<g2, 256, 0, stream>>>(hmid, w2bf, b2, row_base, list_dest, ybuf);

  moe_combine<<<((size_t)NTOK * HDIM) / 256, 256, 0, stream>>>(ybuf, top_w, out);
}
